// convex_upsample_layer_5463198400671
// MI455X (gfx1250) — matrix operand order repaired and checked by reading, not compiled, not run on hardware
//
#include <hip/hip_runtime.h>

typedef __attribute__((ext_vector_type(16))) __bf16 v16bf;
typedef __attribute__((ext_vector_type(8)))  float  v8f;
typedef __attribute__((ext_vector_type(8)))  unsigned short ushort8;
typedef __attribute__((ext_vector_type(4)))  unsigned uintx4;
typedef __attribute__((ext_vector_type(8)))  int intx8;
typedef __attribute__((ext_vector_type(4)))  int intx4;

union BF16Frag { ushort8 h[2]; v16bf v; };

#if __has_builtin(__builtin_amdgcn_tensor_load_to_lds) && __has_builtin(__builtin_amdgcn_s_wait_tensorcnt)
#define HAVE_TDM 1
#else
#define HAVE_TDM 0
#endif

__device__ __forceinline__ unsigned short f2bf(float x) {
    unsigned u = __builtin_bit_cast(unsigned, x);
    unsigned r = (u + 0x7FFFu + ((u >> 16) & 1u)) >> 16;  // round-to-nearest-even
    return (unsigned short)r;
}

#define B_   8
#define C_   64
#define H_   128
#define W_   416
#define TW   16      // pixels per row-tile (WMMA M)
#define TR   8       // rows per workgroup (one per wave)
#define KSTEPS1 18   // conv3x3: K = 9*64 = 576, 32 per WMMA
#define NT1     4    // 64 hidden channels / 16
#define KSTEPS2 2    // conv1x1: K = 64
#define NT2     9    // 144 mask channels / 16  (tile g == softmax tap g)

#define W1FRAG_DW (KSTEPS1*NT1*256)   // 18432 dwords = 72 KB
#define W2FRAG_DW (NT2*KSTEPS2*256)   //  4608 dwords = 18 KB

// ---- weight pre-pack: f32 OIHW -> bf16 in per-lane WMMA B-fragment order ----
// B layout this code uses (32x16 KxN bf16, same K order as A): lane l<16 -> N=l,  K=0..15 contiguous;
//                                     lane l>=16 -> N=l-16, K=16..31 contiguous.
__global__ void pack_weights(const float* __restrict__ w1,
                             const float* __restrict__ w2,
                             unsigned* __restrict__ w1f,
                             unsigned* __restrict__ w2f) {
    int tid = blockIdx.x * blockDim.x + threadIdx.x;
    if (tid < W1FRAG_DW) {
        int f = tid >> 8, lane = (tid >> 3) & 31, v = tid & 7;
        int kt = f >> 2, nt = f & 3;
        int n  = nt*16 + (lane & 15);
        int kb = (lane < 16) ? 0 : 16;
        int k0 = kt*32 + kb + 2*v, k1 = k0 + 1;     // k = tap*64 + c
        int tap0 = k0 >> 6, c0 = k0 & 63;
        int tap1 = k1 >> 6, c1 = k1 & 63;
        float a = w1[((n*64 + c0)*3 + tap0/3)*3 + (tap0%3)];
        float b = w1[((n*64 + c1)*3 + tap1/3)*3 + (tap1%3)];
        w1f[tid] = (unsigned)f2bf(a) | ((unsigned)f2bf(b) << 16);
    } else if (tid < W1FRAG_DW + W2FRAG_DW) {
        int t = tid - W1FRAG_DW;
        int f = t >> 8, lane = (t >> 3) & 31, v = t & 7;
        int g = f >> 1, kt = f & 1;
        int n  = g*16 + (lane & 15);
        int kb = (lane < 16) ? 0 : 16;
        int k0 = kt*32 + kb + 2*v, k1 = k0 + 1;     // k = input channel
        float a = w2[n*64 + k0];
        float b = w2[n*64 + k1];
        w2f[t] = (unsigned)f2bf(a) | ((unsigned)f2bf(b) << 16);
    }
}

// TDM 2D tile load: Group0/Group1 descriptor per CDNA5 ISA §8.3/§8.4.
// tensor_d* are the remaining extents measured from the tile start (OOB -> zero).
// This toolchain exposes the 6-arg builtin: (g0, g1, g2, g3, g4, cpol).
__device__ __forceinline__ void tdm_load_2d(unsigned lds_off, const void* gaddr,
                                            unsigned tensor_d0, unsigned tensor_d1,
                                            unsigned tile_d0, unsigned tile_d1,
                                            unsigned stride0_elems, unsigned data_size_sel /*0:1B 1:2B 2:4B 3:8B*/)
{
#if HAVE_TDM
    unsigned long long ga = (unsigned long long)gaddr;
    uintx4 g0;
    g0[0] = 1u;                                   // count=1, user descriptor
    g0[1] = lds_off;                              // lds_addr (bytes)
    g0[2] = (unsigned)ga;                         // global_addr[31:0]
    g0[3] = (unsigned)(ga >> 32) | 0x80000000u;   // global_addr[56:32] | type=2 (bits 127:126)
    intx8 g1;
    g1[0] = (int)(data_size_sel << 16);           // wg_mask=0 | data_size
    g1[1] = (int)((tensor_d0 & 0xFFFFu) << 16);   // tensor_dim0[15:0]  (bits 63:48)
    g1[2] = (int)((tensor_d0 >> 16) | ((tensor_d1 & 0xFFFFu) << 16)); // dim0 hi | dim1 lo
    g1[3] = (int)((tensor_d1 >> 16) | (tile_d0 << 16));               // dim1 hi | tile_dim0
    g1[4] = (int)(tile_d1 & 0xFFFFu);             // tile_dim1 | tile_dim2=0
    g1[5] = (int)stride0_elems;                   // tensor_dim0_stride[31:0]
    g1[6] = 0;                                    // stride0 hi | dim1_stride lo (unused, 2D)
    g1[7] = 0;
    intx4 z4 = {0, 0, 0, 0};
    intx8 z8 = {0, 0, 0, 0, 0, 0, 0, 0};
    __builtin_amdgcn_tensor_load_to_lds(g0, g1, z4, z4, z8, 0);
#else
    (void)lds_off; (void)gaddr; (void)tensor_d0; (void)tensor_d1;
    (void)tile_d0; (void)tile_d1; (void)stride0_elems; (void)data_size_sel;
#endif
}

// ---- fused: conv3x3+relu -> conv1x1 -> softmax(9) -> convex combine -> shuffle ----
__global__ __launch_bounds__(256, 4) void convex_upsample_fused(
    const float* __restrict__ depth,
    const float* __restrict__ feat,
    const unsigned short* __restrict__ w1f,
    const unsigned short* __restrict__ w2f,
    float* __restrict__ out)
{
    __shared__ unsigned short featS[10*18*64]; // bf16 feat tile (+halo), 23040 B
    __shared__ unsigned short hS[TR*16*64];    // bf16 hidden tile, wave-private, 16384 B
    __shared__ unsigned short w2S[W2FRAG_DW*2];// packed w2 fragments, 18432 B
    __shared__ float depthS[10*18];            // depth tile (+halo), 720 B

    const int tid  = threadIdx.x;
    const int wave = tid >> 5;
    const int lane = tid & 31;
    const int b  = blockIdx.z;
    const int y0 = blockIdx.y * TR;
    const int x0 = blockIdx.x * TW;

    // stage feat (rows y0-1..y0+8, cols x0-1..x0+16) as [r][x][c] bf16, zero-padded
    for (int d = tid; d < 10*18*32; d += 256) {
        int r  = d / 576;
        int xl = (d % 576) >> 5;
        int c0 = (d & 31) * 2;
        int y = y0 - 1 + r, x = x0 - 1 + xl;
        unsigned lo = 0, hi = 0;
        if (y >= 0 && y < H_ && x >= 0 && x < W_) {
            const float* p = feat + (((b*C_ + c0)*H_ + y)*W_ + x);
            lo = f2bf(p[0]);
            hi = f2bf(p[H_*W_]);
        }
        ((unsigned*)featS)[d] = lo | (hi << 16);
    }

    const bool interior = (x0 > 0) && (y0 > 0);
#if HAVE_TDM
    // TDM: DMA the w2 fragment block (1D) and, for interior tiles, the 18x10
    // depth tile (2D, right/bottom OOB auto-zeroed via tensor dims).
    if (wave == 0) {
        tdm_load_2d((unsigned)(unsigned long long)w2S, w2f,
                    W2FRAG_DW, 1, W2FRAG_DW, 1, W2FRAG_DW, 2u);
        if (interior) {
            const float* dg = depth + ((b*H_ + (y0 - 1))*W_ + (x0 - 1));
            tdm_load_2d((unsigned)(unsigned long long)depthS, dg,
                        (unsigned)(W_ - (x0 - 1)), (unsigned)(H_ - (y0 - 1)),
                        18u, 10u, (unsigned)W_, 2u);
        }
        __builtin_amdgcn_s_wait_tensorcnt(0);
    }
    if (!interior) {
        for (int d = tid; d < 180; d += 256) {
            int r = d / 18, xl = d % 18;
            int y = y0 - 1 + r, x = x0 - 1 + xl;
            depthS[d] = (y >= 0 && y < H_ && x >= 0 && x < W_)
                            ? depth[(b*H_ + y)*W_ + x] : 0.f;
        }
    }
#else
    for (int d = tid; d < 180; d += 256) {
        int r = d / 18, xl = d % 18;
        int y = y0 - 1 + r, x = x0 - 1 + xl;
        depthS[d] = (y >= 0 && y < H_ && x >= 0 && x < W_)
                        ? depth[(b*H_ + y)*W_ + x] : 0.f;
    }
    for (int d = tid; d < W2FRAG_DW; d += 256)
        ((unsigned*)w2S)[d] = ((const unsigned*)w2f)[d];
#endif
    __builtin_prefetch(w1f + tid*144, 0, 0);   // warm w1 fragments toward L2
    __syncthreads();

    const int m    = lane & 15;               // A-matrix row (pixel in tile)
    const int ka0  = (lane < 16) ? 0 : 16;    // A-fragment K-run start (contiguous K per lane half)
    const int moff = (lane < 16) ? 0 : 8;     // C-matrix M offset per lane half

    // ---- phase 1: conv3x3 via implicit GEMM (bf16 WMMA, f32 accum) ----
    v8f acc[NT1] = {};
    for (int kt = 0; kt < KSTEPS1; ++kt) {
        const int tap = kt >> 1;
        const int ky = tap / 3, kx = tap % 3;
        const int chalf = (kt & 1) * 32;
        const int base = ((wave + ky)*18 + (m + kx))*64 + chalf + ka0;
        BF16Frag a;
        a.h[0] = *(const ushort8*)(featS + base);        // K run0 (8 vals)
        a.h[1] = *(const ushort8*)(featS + base + 8);    // K run1 (8 vals)
        #pragma unroll
        for (int nt = 0; nt < NT1; ++nt) {
            const unsigned short* wp = w1f + ((kt*NT1 + nt)*512 + lane*16);
            BF16Frag bm;
            bm.h[0] = *(const ushort8*)(wp);
            bm.h[1] = *(const ushort8*)(wp + 8);
            acc[nt] = __builtin_amdgcn_wmma_f32_16x16x32_bf16(
                false, a.v, false, bm.v, (short)0, acc[nt], false, false);
        }
    }

    // relu -> bf16 into wave-private LDS h tile, layout [wave][m][c]
    #pragma unroll
    for (int nt = 0; nt < NT1; ++nt) {
        #pragma unroll
        for (int rr = 0; rr < 8; ++rr) {
            float v = acc[nt][rr]; v = v > 0.f ? v : 0.f;
            hS[(wave*16 + rr + moff)*64 + nt*16 + (lane & 15)] = f2bf(v);
        }
    }
    // wave-private region + same-wave DS ordering: no barrier needed

    // ---- phase 2: conv1x1 -> 144 mask logits (tile g == tap g, N == subpixel) ----
    v8f macc[NT2] = {};
    #pragma unroll
    for (int kt = 0; kt < KSTEPS2; ++kt) {
        const int base = (wave*16 + m)*64 + kt*32 + ka0;
        BF16Frag a;
        a.h[0] = *(const ushort8*)(hS + base);
        a.h[1] = *(const ushort8*)(hS + base + 8);
        #pragma unroll
        for (int g = 0; g < NT2; ++g) {
            const unsigned short* wp = w2S + ((g*KSTEPS2 + kt)*512 + lane*16);
            BF16Frag bm;
            bm.h[0] = *(const ushort8*)(wp);
            bm.h[1] = *(const ushort8*)(wp + 8);
            macc[g] = __builtin_amdgcn_wmma_f32_16x16x32_bf16(
                false, a.v, false, bm.v, (short)0, macc[g], false, false);
        }
    }

    // ---- phase 3: per-lane softmax over 9 taps, convex combine, shuffled store ----
    const int s  = lane & 15;          // subpixel = WMMA N index
    const int s1 = s >> 2, s2 = s & 3;
    const int y  = y0 + wave;
    #pragma unroll
    for (int rr = 0; rr < 8; ++rr) {
        const int mm = rr + moff;      // pixel within row tile
        const int x  = x0 + mm;
        float mx = macc[0][rr];
        #pragma unroll
        for (int g = 1; g < 9; ++g) mx = fmaxf(mx, macc[g][rr]);
        float num = 0.f, den = 0.f;
        #pragma unroll
        for (int g = 0; g < 9; ++g) {
            const int ky = g / 3, kx = g % 3;
            float e = __expf(macc[g][rr] - mx);
            den += e;
            num += e * depthS[(wave + ky)*18 + (mm + kx)];
        }
        out[(b*(H_*4) + 4*y + s1)*(W_*4) + 4*x + s2] = num / den;
    }
}

extern "C" void kernel_launch(void* const* d_in, const int* in_sizes, int n_in,
                              void* d_out, int out_size, void* d_ws, size_t ws_size,
                              hipStream_t stream) {
    const float* depth = (const float*)d_in[0];
    const float* feat  = (const float*)d_in[1];
    const float* w1    = (const float*)d_in[2];
    const float* w2    = (const float*)d_in[3];
    float* out = (float*)d_out;

    unsigned* w1f = (unsigned*)d_ws;           // 73728 B
    unsigned* w2f = w1f + W1FRAG_DW;           // 18432 B (total ws use: 92160 B)

    const int packN = W1FRAG_DW + W2FRAG_DW;   // 23040 threads
    pack_weights<<<(packN + 255)/256, 256, 0, stream>>>(w1, w2, w1f, w2f);

    dim3 grid(W_/TW, H_/TR, B_);               // 26 x 16 x 8
    convex_upsample_fused<<<grid, 256, 0, stream>>>(
        depth, feat, (const unsigned short*)w1f, (const unsigned short*)w2f, out);
}
